// PredictiveCodingNet_51041391345769
// MI455X (gfx1250) — matrix operand order repaired and checked by reading, not compiled, not run on hardware
//
#include <hip/hip_runtime.h>

// Problem constants (match reference)
#define BATCH  4096
#define IN_    1024
#define HID_   2048
#define OUT_   1000
#define OUTP   1024   // OUT padded to multiple of 128 for WMMA tiling
#define ITERS_ 50
#define LR_    0.001f

typedef __bf16 bf16;
typedef __attribute__((ext_vector_type(16))) __bf16 v16bf;
typedef __attribute__((ext_vector_type(8)))  float  v8f;

union FragU { v16bf v; uint4 u[2]; };

// LDS row stride in bf16 elements: 40 elems = 80 bytes = 20 banks.
// 20*r mod 64 is a permutation over r=0..15 -> conflict-free b128 reads.
constexpr int LDSS = 40;
constexpr int BUFO = 128 * LDSS;   // double-buffer offset (bf16 elems)

// ---------------------------------------------------------------------------
// Shared WMMA mainloop: C[128x128 tile] = A[M,K] @ Bt[N,K]^T (Bt is [N][K]).
// 256 threads = 8 waves (4x2); wave tile 32x64 = 2x4 WMMA 16x16 tiles.
// LDS double-buffered: one barrier per K-chunk; next chunk's global loads
// are staged in VGPRs while the current chunk's WMMAs execute.
// ---------------------------------------------------------------------------
__device__ __forceinline__ void wmma_mainloop(
    const bf16* __restrict__ A, const bf16* __restrict__ Bt, int K,
    int m0, int n0, bf16* As, bf16* Bs, v8f acc[2][4])
{
  const int tid  = threadIdx.x;
  const int lane = tid & 31;
  const int w    = tid >> 5;
  const int wm   = w >> 1;          // 0..3 : wave row (32 rows each)
  const int wn   = w & 1;           // 0..1 : wave col (64 cols each)
  const int g    = lane >> 4;       // lane group (0/1)
  const int r    = lane & 15;       // row/col within 16
  const int lrow = tid >> 2;        // 0..63 : LDS load row
  const int lcol = (tid & 3) << 3;  // 0,8,16,24 : LDS load col (bf16 units)
  const int st   = lrow * LDSS + lcol;

  // Hoisted global base pointers (advance via loop offset only)
  const bf16* gA0 = A  + (size_t)(m0 + lrow     ) * K + lcol;
  const bf16* gA1 = A  + (size_t)(m0 + lrow + 64) * K + lcol;
  const bf16* gB0 = Bt + (size_t)(n0 + lrow     ) * K + lcol;
  const bf16* gB1 = Bt + (size_t)(n0 + lrow + 64) * K + lcol;

  const v8f zero = {0.f,0.f,0.f,0.f,0.f,0.f,0.f,0.f};
#pragma unroll
  for (int i = 0; i < 2; ++i)
#pragma unroll
    for (int j = 0; j < 4; ++j)
      acc[i][j] = zero;

  // Prologue: stage chunk 0 into buffer 0
  {
    const uint4 a0 = *(const uint4*)gA0;
    const uint4 a1 = *(const uint4*)gA1;
    const uint4 b0 = *(const uint4*)gB0;
    const uint4 b1 = *(const uint4*)gB1;
    *(uint4*)(As + st) = a0; *(uint4*)(As + st + 64 * LDSS) = a1;
    *(uint4*)(Bs + st) = b0; *(uint4*)(Bs + st + 64 * LDSS) = b1;
  }

  int buf = 0;
  for (int k0 = 0; k0 < K; k0 += 32) {
    // One barrier per chunk: makes buffer `buf` visible AND guarantees all
    // waves' reads of buffer buf^1 (previous chunk) completed (dscnt==0).
    __syncthreads();

    const bool more = (k0 + 32 < K);
    uint4 na0, na1, nb0, nb1;
    if (more) {   // stage next chunk while WMMAs run
      na0 = *(const uint4*)(gA0 + k0 + 32);
      na1 = *(const uint4*)(gA1 + k0 + 32);
      nb0 = *(const uint4*)(gB0 + k0 + 32);
      nb1 = *(const uint4*)(gB1 + k0 + 32);
    }

    const bf16* Ab = As + buf * BUFO;
    const bf16* Bb = Bs + buf * BUFO;

    // A fragment (16x32 bf16): lanes 0-15 -> K {0..7,16..23}, 16-31 -> {8..15,24..31}
    FragU aF[2];
#pragma unroll
    for (int i = 0; i < 2; ++i) {
      const bf16* p = Ab + (wm * 32 + i * 16 + r) * LDSS + g * 8;
      aF[i].u[0] = *(const uint4*)(p);
      aF[i].u[1] = *(const uint4*)(p + 16);
    }
    // B fragment (32x16 bf16 from [N][K] tile): lanes 0-15 -> K {0..7,16..23}, 16-31 -> {8..15,24..31}
    FragU bF[4];
#pragma unroll
    for (int j = 0; j < 4; ++j) {
      const bf16* p = Bb + (wn * 64 + j * 16 + r) * LDSS + g * 8;
      bF[j].u[0] = *(const uint4*)(p);
      bF[j].u[1] = *(const uint4*)(p + 16);
    }
#pragma unroll
    for (int i = 0; i < 2; ++i)
#pragma unroll
      for (int j = 0; j < 4; ++j)
        acc[i][j] = __builtin_amdgcn_wmma_f32_16x16x32_bf16(
            false, aF[i].v, false, bF[j].v, (short)0, acc[i][j], false, false);

    if (more) {   // commit next chunk to the alternate buffer
      bf16* An = As + (buf ^ 1) * BUFO;
      bf16* Bn = Bs + (buf ^ 1) * BUFO;
      *(uint4*)(An + st) = na0; *(uint4*)(An + st + 64 * LDSS) = na1;
      *(uint4*)(Bn + st) = nb0; *(uint4*)(Bn + st + 64 * LDSS) = nb1;
    }
    buf ^= 1;
  }
}

__device__ __forceinline__ void block_reduce_add(float v, float* red, float* dst)
{
  const int tid = threadIdx.x;
  red[tid] = v;
  __syncthreads();
#pragma unroll
  for (int s = 128; s > 0; s >>= 1) {
    if (tid < s) red[tid] += red[tid + s];
    __syncthreads();
  }
  if (tid == 0) atomicAdd(dst, red[0]);
}

// ---------------------------------------------------------------------------
// Step 1: x_pred = sigmoid(h @ W_rec + bias_gen); e_x = x - x_pred (bf16);
//         energy += sum(e_x^2)
// ---------------------------------------------------------------------------
__global__ __launch_bounds__(256) void k_step1(
    const bf16* __restrict__ hb, const bf16* __restrict__ WrecT,
    const float* __restrict__ x, const float* __restrict__ bias_gen,
    bf16* __restrict__ ex, float* __restrict__ energy)
{
  __shared__ __attribute__((aligned(16))) bf16 As[2 * BUFO];
  __shared__ __attribute__((aligned(16))) bf16 Bs[2 * BUFO];
  __shared__ float red[256];
  v8f acc[2][4];
  const int m0 = blockIdx.y * 128, n0 = blockIdx.x * 128;
  wmma_mainloop(hb, WrecT, HID_, m0, n0, As, Bs, acc);

  const int tid = threadIdx.x, lane = tid & 31, w = tid >> 5;
  const int wm = w >> 1, wn = w & 1, g = lane >> 4, r = lane & 15;
  float le = 0.f;
#pragma unroll
  for (int i = 0; i < 2; ++i)
#pragma unroll
    for (int j = 0; j < 4; ++j) {
      const int n  = n0 + wn * 64 + j * 16 + r;
      const int mb = m0 + wm * 32 + i * 16 + g * 8;
      const float bg = bias_gen[n];
#pragma unroll
      for (int v = 0; v < 8; ++v) {
        const int m = mb + v;
        const float s = acc[i][j][v] + bg;
        // sigmoid via single v_exp_f32 + single v_rcp_f32 (no IEEE div chain)
        const float p = __builtin_amdgcn_rcpf(1.0f + __expf(-s));
        const float e = x[(size_t)m * IN_ + n] - p;
        ex[(size_t)m * IN_ + n] = (bf16)e;
        le += e * e;
      }
    }
  block_reduce_add(le, red, energy);
}

// ---------------------------------------------------------------------------
// Step 2: h += LR * (e_x @ W_rec^T)   (f32 master + bf16 copy)
// ---------------------------------------------------------------------------
__global__ __launch_bounds__(256) void k_step2(
    const bf16* __restrict__ ex, const bf16* __restrict__ Wrec,
    float* __restrict__ h32, bf16* __restrict__ hb)
{
  __shared__ __attribute__((aligned(16))) bf16 As[2 * BUFO];
  __shared__ __attribute__((aligned(16))) bf16 Bs[2 * BUFO];
  v8f acc[2][4];
  const int m0 = blockIdx.y * 128, n0 = blockIdx.x * 128;
  wmma_mainloop(ex, Wrec, IN_, m0, n0, As, Bs, acc);

  const int tid = threadIdx.x, lane = tid & 31, w = tid >> 5;
  const int wm = w >> 1, wn = w & 1, g = lane >> 4, r = lane & 15;
#pragma unroll
  for (int i = 0; i < 2; ++i)
#pragma unroll
    for (int j = 0; j < 4; ++j) {
      const int n  = n0 + wn * 64 + j * 16 + r;
      const int mb = m0 + wm * 32 + i * 16 + g * 8;
#pragma unroll
      for (int v = 0; v < 8; ++v) {
        const size_t idx = (size_t)(mb + v) * HID_ + n;
        const float hn = h32[idx] + LR_ * acc[i][j][v];
        h32[idx] = hn;
        hb[idx]  = (bf16)hn;
      }
    }
}

// ---------------------------------------------------------------------------
// Step 3: y_pred = h @ W_out^T + bias; e_y = y - y_pred; y += LR*e_y;
//         energy += sum(e_y^2). N padded to 1024 (pad cols stay exactly 0).
// ---------------------------------------------------------------------------
__global__ __launch_bounds__(256) void k_step3(
    const bf16* __restrict__ hb, const bf16* __restrict__ WoutT,
    const float* __restrict__ bias_pad, float* __restrict__ y32,
    float* __restrict__ energy)
{
  __shared__ __attribute__((aligned(16))) bf16 As[2 * BUFO];
  __shared__ __attribute__((aligned(16))) bf16 Bs[2 * BUFO];
  __shared__ float red[256];
  v8f acc[2][4];
  const int m0 = blockIdx.y * 128, n0 = blockIdx.x * 128;
  wmma_mainloop(hb, WoutT, HID_, m0, n0, As, Bs, acc);

  const int tid = threadIdx.x, lane = tid & 31, w = tid >> 5;
  const int wm = w >> 1, wn = w & 1, g = lane >> 4, r = lane & 15;
  float le = 0.f;
#pragma unroll
  for (int i = 0; i < 2; ++i)
#pragma unroll
    for (int j = 0; j < 4; ++j) {
      const int n  = n0 + wn * 64 + j * 16 + r;
      const int mb = m0 + wm * 32 + i * 16 + g * 8;
      const float bp = bias_pad[n];
#pragma unroll
      for (int v = 0; v < 8; ++v) {
        const size_t idx = (size_t)(mb + v) * OUTP + n;
        const float yp = acc[i][j][v] + bp;
        const float yv = y32[idx];
        const float e  = yv - yp;
        y32[idx] = yv + LR_ * e;
        le += e * e;
      }
    }
  block_reduce_add(le, red, energy);
}

// ---------------------------------------------------------------------------
// Prep / init / finalize
// ---------------------------------------------------------------------------
__global__ __launch_bounds__(256) void k_prep_wrec(
    const float* __restrict__ W, bf16* __restrict__ Wb, bf16* __restrict__ WbT)
{
  const int i = blockIdx.x * 256 + threadIdx.x;
  if (i < HID_ * IN_) {
    const int kk = i / IN_, n = i % IN_;
    const float v = W[i];
    Wb[i] = (bf16)v;                          // [HID][IN]  (Bt for GEMM2)
    WbT[(size_t)n * HID_ + kk] = (bf16)v;     // [IN][HID]  (Bt for GEMM1)
  }
}

__global__ __launch_bounds__(256) void k_prep_wout(
    const float* __restrict__ W, const float* __restrict__ b,
    bf16* __restrict__ WT, float* __restrict__ bpad)
{
  const int i = blockIdx.x * 256 + threadIdx.x;
  if (i < OUTP * HID_) {
    const int n = i / HID_, kk = i % HID_;
    WT[i] = (n < OUT_) ? (bf16)W[(size_t)n * HID_ + kk] : (bf16)0.f;
  }
  if (i < OUTP) bpad[i] = (i < OUT_) ? b[i] : 0.f;
}

__global__ __launch_bounds__(256) void k_zero(
    float* __restrict__ h32, bf16* __restrict__ hb,
    float* __restrict__ y32, float* __restrict__ energy)
{
  const size_t i = (size_t)blockIdx.x * 256 + threadIdx.x;
  if (i < (size_t)BATCH * HID_) { h32[i] = 0.f; hb[i] = (bf16)0.f; }
  if (i < (size_t)BATCH * OUTP) y32[i] = 0.f;
  if (i == 0) *energy = 0.f;
}

__global__ __launch_bounds__(256) void k_final(
    const float* __restrict__ y32, const float* __restrict__ energy,
    float* __restrict__ out)
{
  const size_t i = (size_t)blockIdx.x * 256 + threadIdx.x;
  const size_t ny = (size_t)BATCH * OUT_;
  if (i < ny) {
    const size_t bi = i / OUT_, n = i % OUT_;
    out[i] = y32[bi * OUTP + n];
  } else if (i == ny) {
    out[i] = (*energy) / (float)(BATCH * ITERS_);
  }
}

// ---------------------------------------------------------------------------
extern "C" void kernel_launch(void* const* d_in, const int* in_sizes, int n_in,
                              void* d_out, int out_size, void* d_ws, size_t ws_size,
                              hipStream_t stream)
{
  const float* x        = (const float*)d_in[0];
  const float* W_rec    = (const float*)d_in[1];
  const float* W_out    = (const float*)d_in[2];
  const float* bias_out = (const float*)d_in[3];
  const float* bias_gen = (const float*)d_in[4];

  // Workspace carve-up (~88 MB total)
  char* p = (char*)d_ws;
  bf16*  Wrec_b  = (bf16*)p;  p += (size_t)HID_ * IN_  * 2;  // 4 MB  [HID][IN]
  bf16*  WrecT_b = (bf16*)p;  p += (size_t)HID_ * IN_  * 2;  // 4 MB  [IN][HID]
  bf16*  WoutT_b = (bf16*)p;  p += (size_t)OUTP * HID_ * 2;  // 4 MB  [OUTP][HID]
  float* h32     = (float*)p; p += (size_t)BATCH * HID_ * 4; // 32 MB
  bf16*  hb      = (bf16*)p;  p += (size_t)BATCH * HID_ * 2; // 16 MB
  bf16*  ex      = (bf16*)p;  p += (size_t)BATCH * IN_  * 2; // 8 MB
  float* y32     = (float*)p; p += (size_t)BATCH * OUTP * 4; // 16 MB
  float* bias_p  = (float*)p; p += (size_t)OUTP * 4;
  float* energy  = (float*)p;

  k_prep_wrec<<<(HID_ * IN_ + 255) / 256, 256, 0, stream>>>(W_rec, Wrec_b, WrecT_b);
  k_prep_wout<<<(OUTP * HID_ + 255) / 256, 256, 0, stream>>>(W_out, bias_out, WoutT_b, bias_p);
  k_zero<<<(unsigned)(((size_t)BATCH * HID_ + 255) / 256), 256, 0, stream>>>(h32, hb, y32, energy);

  const dim3 blk(256);
  const dim3 g1(IN_  / 128, BATCH / 128);  //  8 x 32
  const dim3 g2(HID_ / 128, BATCH / 128);  // 16 x 32
  const dim3 g3(OUTP / 128, BATCH / 128);  //  8 x 32
  for (int it = 0; it < ITERS_; ++it) {
    k_step1<<<g1, blk, 0, stream>>>(hb, WrecT_b, x, bias_gen, ex, energy);
    k_step2<<<g2, blk, 0, stream>>>(ex, Wrec_b, h32, hb);
    k_step3<<<g3, blk, 0, stream>>>(hb, WoutT_b, bias_p, y32, energy);
  }

  const size_t nfin = (size_t)BATCH * OUT_ + 1;
  k_final<<<(unsigned)((nfin + 255) / 256), 256, 0, stream>>>(y32, energy, (float*)d_out);
}
